// DecodeLayer_56650618634692
// MI455X (gfx1250) — compile-verified, matrix operand order checked by reading, not run on hardware
//
// MI455X (gfx1250) detection post-processing: decode + softmax-max + NMS.
//
// Phase 1 (decode_kernel): bandwidth-bound pass over logits (267 MB -> ~11.5us
//   roofline at 23.3 TB/s). Stages 128 anchors x 85 f32 per block into LDS via
//   CDNA5 async-to-LDS (global_load_async_to_lds_b128 + s_wait_asynccnt, with
//   fallback), computes class max/argmax, and the softmax denominator via
//   v_wmma_f32_16x16x32_f16 as a (16x96 exp-matrix) x (ones e0-column) product
//   (f32 accumulate). score = 1/sum; argmax(softmax)==argmax(logits).
// Phase 2 (nms_kernel): one 1024-thread WG per image. Uses the 320KB WGP LDS:
//   24576 f32 scores (98KB) + 24576 boxes as 4xf16 (196KB) resident in LDS.
//   200 sequential steps, each = 2-barrier block argmax (local maxima kept
//   incrementally) + fused IoU-suppress pass. Early-exit when max <= NEG/2
//   (scores are monotone non-increasing, so all later keeps are false).
//
// Workspace requirement: B*N*(4+1+1)*4 = 18,865,152 bytes.

#include <hip/hip_runtime.h>
#include <hip/hip_fp16.h>

#define B_IMGS   32
#define N_ANCH   24564
#define N_CLS    81
#define LSTRIDE  85          // 4 loc + 81 class logits
#define MAXDET   200
#define IOU_THR  0.45f
#define CONF_THR 0.01f
#define NEGV     (-1000000000.0f)
#define NEG_HALF (-500000000.0f)

#define APB         128      // anchors per decode block (8 waves x 16 rows)
#define DEC_THREADS 256
#define NMS_THREADS 1024
#define NPAD        24576    // N_ANCH rounded up to NMS_THREADS multiple
#define ITEMS       (NPAD / NMS_THREADS)   // 24

static_assert(ITEMS * NMS_THREADS == NPAD, "padding mismatch");

typedef __attribute__((ext_vector_type(16))) _Float16 v16h;
typedef __attribute__((ext_vector_type(8)))  float    v8f;

// ---- CDNA5 async-to-LDS path (guarded; falls back to plain vector copy) ----
#if defined(__has_builtin)
#  if __has_builtin(__builtin_amdgcn_global_load_async_to_lds_b128)
#    define HAVE_ASYNC_B128 1
#  endif
#  if __has_builtin(__builtin_amdgcn_s_wait_asynccnt)
#    define HAVE_WAIT_ASYNC 1
#  endif
#endif

// Builtin parameter types, exactly as the compiler declares them:
//   (__device__ v4i*, __shared__ v4i*, imm int, imm int)
typedef int v4i __attribute__((vector_size(16)));
typedef __attribute__((address_space(1))) v4i gv4i;   // global (__device__)
typedef __attribute__((address_space(3))) v4i lv4i;   // LDS (__shared__)

__device__ __forceinline__ gv4i* to_glb_v4(const void* p) {
  return (gv4i*)(unsigned long long)p;
}
__device__ __forceinline__ lv4i* to_lds_v4(void* p) {
  // Flat LDS addresses are {aperture[63:32], offset[31:0]}; truncation yields
  // the LDS byte offset, which is what an AS(3) pointer holds.
  return (lv4i*)(unsigned)(unsigned long long)p;
}

__device__ __forceinline__ void wait_async0() {
#if defined(HAVE_WAIT_ASYNC)
  __builtin_amdgcn_s_wait_asynccnt(0);
#elif defined(HAVE_ASYNC_B128)
  asm volatile("s_wait_asynccnt 0" ::: "memory");
#endif
}

// ============================ Phase 1: decode ===============================
__global__ __launch_bounds__(DEC_THREADS) void decode_kernel(
    const float* __restrict__ logits,   // [B, N, 85]
    const float* __restrict__ anchors,  // [N, 4]
    float* __restrict__ wsBoxes,        // [B, N, 4]
    float* __restrict__ wsScores,       // [B, N] (masked: NEGV if invalid)
    int*   __restrict__ wsClasses)      // [B, N]
{
  __shared__ __align__(16) float slog[APB * LSTRIDE];  // 43,520 B
  __shared__ float ldsSum[APB];

  const int bimg = blockIdx.y;
  const int blockStart = blockIdx.x * APB;
  const size_t baseFloat = ((size_t)bimg * N_ANCH + blockStart) * LSTRIDE;
  const float* src = logits + baseFloat;
  const size_t totalFloats = (size_t)B_IMGS * N_ANCH * LSTRIDE;
  const int nChunks = APB * LSTRIDE / 4;  // 2720 x b128
  const int tid = threadIdx.x;

#if defined(HAVE_ASYNC_B128)
  for (int c = tid; c < nChunks; c += DEC_THREADS) {
    if (baseFloat + (size_t)(c * 4) + 4 <= totalFloats) {
      __builtin_amdgcn_global_load_async_to_lds_b128(
          to_glb_v4(src + c * 4), to_lds_v4(&slog[c * 4]), 0, 0);
    }
  }
  wait_async0();
#else
  for (int c = tid; c < nChunks; c += DEC_THREADS) {
    if (baseFloat + (size_t)(c * 4) + 4 <= totalFloats) {
      float4 v = *(const float4*)(src + c * 4);
      *(float4*)(&slog[c * 4]) = v;
    }
  }
#endif
  __syncthreads();

  const int lane = tid & 31;
  const int wave = tid >> 5;
  const int a = lane & 15;   // matrix row (anchor within wave tile)
  const int h = lane >> 4;   // which half of the K range this lane owns
  const int aLoc = wave * 16 + a;
  const int g = blockStart + aLoc;
  const float* cl = &slog[aLoc * LSTRIDE];

  // --- class max/argmax over this lane's subset (ascending class order) ---
  float m = -3.0e38f; int cls = 0;
#pragma unroll
  for (int q = 0; q < 3; ++q) {
#pragma unroll
    for (int j = 0; j < 16; ++j) {
      const int k = h * 8 + j + ((j >= 8) ? 8 : 0);  // ISA A-layout K mapping
      const int c = q * 32 + k;
      if (c < N_CLS) {
        float v = cl[4 + c];
        if (v > m) { m = v; cls = c; }
      }
    }
  }
  {  // combine with partner lane (L ^ 16 holds complementary classes)
    float om = __shfl_xor(m, 16, 32);
    int   oc = __shfl_xor(cls, 16, 32);
    if (om > m || (om == m && oc < cls)) { m = om; cls = oc; }
  }

  // --- softmax denominator via WMMA: D[:,0] = A(16x96 exp) x ones ---
  v16h bv;
#pragma unroll
  for (int j = 0; j < 16; ++j)
    bv[j] = (a == 0) ? (_Float16)1.0f : (_Float16)0.0f;  // B[k][0]=1, else 0

  v8f acc = {};
#pragma unroll
  for (int q = 0; q < 3; ++q) {
    v16h av;
#pragma unroll
    for (int j = 0; j < 16; ++j) {
      const int k = h * 8 + j + ((j >= 8) ? 8 : 0);
      const int c = q * 32 + k;
      float e = (c < N_CLS) ? __expf(cl[4 + c] - m) : 0.0f;
      av[j] = (_Float16)e;
    }
    acc = __builtin_amdgcn_wmma_f32_16x16x32_f16(
        false, av, false, bv, (short)0, acc, false, false);
  }
  // D layout: lane 0 holds M=0..7 (vgpr r=M), lane 16 holds M=8..15 (r=M-8).
  if (a == 0) {
#pragma unroll
    for (int r = 0; r < 8; ++r) ldsSum[wave * 16 + h * 8 + r] = acc[r];
  }
  __syncthreads();
  const float ssum = ldsSum[wave * 16 + a];
  const float score = 1.0f / ssum;   // max softmax prob

  if (h == 0 && g < N_ANCH) {
    const bool valid = (cls != 0) && (score >= CONF_THR);
    const float ms = valid ? score : NEGV;

    const float l0 = cl[0], l1 = cl[1], l2 = cl[2], l3 = cl[3];
    const float a0 = anchors[g * 4 + 0], a1 = anchors[g * 4 + 1];
    const float a2 = anchors[g * 4 + 2], a3 = anchors[g * 4 + 3];
    const float w = a2 - a0, hh = a3 - a1;
    const float cx = fmaf(l0, w,  0.5f * (a0 + a2));
    const float cy = fmaf(l1, hh, 0.5f * (a1 + a3));
    const float sx = __expf(l2) * w * 0.5f;
    const float sy = __expf(l3) * hh * 0.5f;
    const float x1 = fminf(fmaxf(cx - sx, 0.0f), 1.0f);
    const float y1 = fminf(fmaxf(cy - sy, 0.0f), 1.0f);
    const float x2 = fminf(fmaxf(cx + sx, 0.0f), 1.0f);
    const float y2 = fminf(fmaxf(cy + sy, 0.0f), 1.0f);

    const size_t o = (size_t)bimg * N_ANCH + g;
    wsBoxes[o * 4 + 0] = x1; wsBoxes[o * 4 + 1] = y1;
    wsBoxes[o * 4 + 2] = x2; wsBoxes[o * 4 + 3] = y2;
    wsScores[o] = ms;
    wsClasses[o] = cls;
  }
}

// ============================ Phase 2: NMS ==================================
__global__ __launch_bounds__(NMS_THREADS) void nms_kernel(
    const float* __restrict__ wsBoxes,
    const float* __restrict__ wsScores,
    const int*   __restrict__ wsClasses,
    float* __restrict__ out)
{
  const int b = blockIdx.x;
  extern __shared__ char smem[];
  float*   sS = (float*)smem;                                 // NPAD f32
  __half2* sB = (__half2*)(smem + (size_t)NPAD * 4);          // NPAD x 2 half2
  float*   sRedV = (float*)(smem + (size_t)NPAD * 4 + (size_t)NPAD * 8);
  int*     sRedI = (int*)(sRedV + 34);

  const int tid = threadIdx.x, lane = tid & 31, wave = tid >> 5;

  // Load scores + boxes (f32 -> packed f16) into LDS; seed per-thread argmax.
  float lmax = -3.0e38f; int lidx = 0;
#pragma unroll
  for (int t = 0; t < ITEMS; ++t) {
    const int i = t * NMS_THREADS + tid;
    float s; float4 bx;
    if (i < N_ANCH) {
      const size_t o = (size_t)b * N_ANCH + i;
      s = wsScores[o];
      const float* bp = wsBoxes + o * 4;
      bx = make_float4(bp[0], bp[1], bp[2], bp[3]);
    } else {
      s = NEGV; bx = make_float4(0.f, 0.f, 0.f, 0.f);
    }
    sS[i] = s;
    sB[2 * i]     = __floats2half2_rn(bx.x, bx.y);
    sB[2 * i + 1] = __floats2half2_rn(bx.z, bx.w);
    if (s > lmax) { lmax = s; lidx = i; }
  }
  __syncthreads();

  float* dBoxes = out;                                  // [B,200,4]
  float* dCls   = out + (size_t)B_IMGS * MAXDET * 4;    // [B,200]
  float* dScr   = dCls + (size_t)B_IMGS * MAXDET;       // [B,200]
  float* dNum   = dScr + (size_t)B_IMGS * MAXDET;       // [B]

  int nkeep = MAXDET;
  for (int step = 0; step < MAXDET; ++step) {
    // ---- block argmax (value desc, index asc on ties = first max) ----
    float v = lmax; int ix = lidx;
#pragma unroll
    for (int off = 16; off > 0; off >>= 1) {
      float ov = __shfl_xor(v, off, 32);
      int   oi = __shfl_xor(ix, off, 32);
      if (ov > v || (ov == v && oi < ix)) { v = ov; ix = oi; }
    }
    if (lane == 0) { sRedV[wave] = v; sRedI[wave] = ix; }
    __syncthreads();
    if (wave == 0) {
      float v2 = sRedV[lane]; int i2 = sRedI[lane];
#pragma unroll
      for (int off = 16; off > 0; off >>= 1) {
        float ov = __shfl_xor(v2, off, 32);
        int   oi = __shfl_xor(i2, off, 32);
        if (ov > v2 || (ov == v2 && oi < i2)) { v2 = ov; i2 = oi; }
      }
      if (lane == 0) { sRedV[32] = v2; sRedI[32] = i2; }
    }
    __syncthreads();
    const float smax = sRedV[32];
    const int imax = sRedI[32];

    if (!(smax > NEG_HALF)) { nkeep = step; break; }  // uniform early exit

    if (tid == 0) {
      const size_t oi = (size_t)b * N_ANCH + imax;
      const size_t od = (size_t)b * MAXDET + step;
      const float* bp = wsBoxes + oi * 4;
      dBoxes[od * 4 + 0] = bp[0]; dBoxes[od * 4 + 1] = bp[1];
      dBoxes[od * 4 + 2] = bp[2]; dBoxes[od * 4 + 3] = bp[3];
      dScr[od] = smax;
      dCls[od] = (float)wsClasses[oi];
    }

    // Broadcast chosen box from LDS (read-only array; no barrier needed).
    const float2 ixy = __half22float2(sB[2 * imax]);
    const float2 izw = __half22float2(sB[2 * imax + 1]);
    const float areaI = (izw.x - ixy.x) * (izw.y - ixy.y);

    // ---- fused IoU suppression + next local argmax (own elements only) ----
    lmax = -3.0e38f; lidx = 0;
#pragma unroll
    for (int t = 0; t < ITEMS; ++t) {
      const int i = t * NMS_THREADS + tid;
      float s = sS[i];
      if (s > NEG_HALF) {
        const float2 oxy = __half22float2(sB[2 * i]);
        const float2 ozw = __half22float2(sB[2 * i + 1]);
        const float lx = fmaxf(ixy.x, oxy.x), ly = fmaxf(ixy.y, oxy.y);
        const float rx = fminf(izw.x, ozw.x), ry = fminf(izw.y, ozw.y);
        const float iw = fmaxf(rx - lx, 0.0f), ih = fmaxf(ry - ly, 0.0f);
        const float inter = iw * ih;
        const float area = (ozw.x - oxy.x) * (ozw.y - oxy.y);
        const float denom = areaI + area - inter + 1e-9f;
        const bool kill = (i == imax) || (inter > IOU_THR * denom);
        if (kill) { s = NEGV; sS[i] = NEGV; }
      }
      if (s > lmax) { lmax = s; lidx = i; }
    }
  }

  // Zero remaining detection slots (keep==false => outputs are zero).
  for (int st = nkeep + tid; st < MAXDET; st += NMS_THREADS) {
    const size_t od = (size_t)b * MAXDET + st;
    dBoxes[od * 4 + 0] = 0.f; dBoxes[od * 4 + 1] = 0.f;
    dBoxes[od * 4 + 2] = 0.f; dBoxes[od * 4 + 3] = 0.f;
    dCls[od] = 0.f; dScr[od] = 0.f;
  }
  if (tid == 0) dNum[b] = (float)nkeep;
}

// ================================ launch ====================================
extern "C" void kernel_launch(void* const* d_in, const int* in_sizes, int n_in,
                              void* d_out, int out_size, void* d_ws, size_t ws_size,
                              hipStream_t stream) {
  const float* logits  = (const float*)d_in[0];  // [B,N,85]
  const float* anchors = (const float*)d_in[1];  // [N,4]
  float* out = (float*)d_out;                    // 38432 floats

  // Workspace partition (requires >= 18,865,152 bytes).
  float* wsBoxes  = (float*)d_ws;                               // B*N*4
  float* wsScores = wsBoxes + (size_t)B_IMGS * N_ANCH * 4;      // B*N
  int*   wsClasses = (int*)(wsScores + (size_t)B_IMGS * N_ANCH); // B*N

  dim3 gDec((N_ANCH + APB - 1) / APB, B_IMGS);
  decode_kernel<<<gDec, DEC_THREADS, 0, stream>>>(
      logits, anchors, wsBoxes, wsScores, wsClasses);

  const size_t smemBytes = (size_t)NPAD * 4 + (size_t)NPAD * 8 + 34 * 4 + 34 * 4;
  nms_kernel<<<dim3(B_IMGS), dim3(NMS_THREADS), smemBytes, stream>>>(
      wsBoxes, wsScores, wsClasses, out);
}
